// NoModulatedConv2d_39573828665626
// MI455X (gfx1250) — compile-verified, matrix operand order checked by reading, not run on hardware
//
#include <hip/hip_runtime.h>
#include <hip/hip_bf16.h>

typedef __attribute__((ext_vector_type(16))) __bf16 v16bf;
typedef __attribute__((ext_vector_type(8)))  __bf16 v8bf;
typedef __attribute__((ext_vector_type(8)))  float  v8f;

#define BATCH 8
#define CIN   512
#define COUT  512
#define H     64
#define W     64
#define ZDIM  512

#define BM   128   // cout tile
#define BN   128   // pixel tile = two image rows
#define KT   32    // WMMA K step (bf16)
#define LDK  40    // padded LDS row stride (elements); 80B rows, 16B-aligned chunks
#define NSLICE (9 * (CIN / KT))   // 144 K-slices of 32

// ---------------------------------------------------------------------------
// Kernel 1: smod[b][cin] = conv_scale * (mod_scale * <style[b], mod_w[cin]> + mod_b[cin])
// ---------------------------------------------------------------------------
__global__ void style_mod_kernel(const float* __restrict__ style,
                                 const float* __restrict__ mod_w,
                                 const float* __restrict__ mod_b,
                                 float* __restrict__ smod) {
    const int t = blockIdx.x * blockDim.x + threadIdx.x;   // 0..4095
    const int b = t >> 9;
    const int i = t & (CIN - 1);
    const float mod_scale  = 0.04419417382415922f;    // 1/sqrt(512)
    const float conv_scale = 0.014731391274719739f;   // 1/sqrt(512*9)
    const float* sv = style + (size_t)b * ZDIM;
    const float* wv = mod_w + (size_t)i * ZDIM;
    float acc = 0.f;
#pragma unroll 8
    for (int z = 0; z < ZDIM; ++z) acc += sv[z] * wv[z];
    smod[t] = conv_scale * (mod_scale * acc + mod_b[i]);
}

// ---------------------------------------------------------------------------
// Kernel 2: implicit-GEMM modulated conv, double-buffered LDS,
// v_wmma_f32_16x16x32_bf16, 8 WMMA tiles per wave per K-slice.
// Block = 256 threads (8 wave32): 128 couts x 128 pixels (2 rows).
// ---------------------------------------------------------------------------
__launch_bounds__(256)
__global__ void modconv_wmma_kernel(const float* __restrict__ input,
                                    const float* __restrict__ weight,
                                    const float* __restrict__ smod,
                                    float* __restrict__ out) {
    __shared__ __bf16 lA[2][BM * LDK];   // modulated weights [m][k], double buffered
    __shared__ __bf16 lB[2][BN * LDK];   // im2col input      [n][k], double buffered
    __shared__ float  lS[CIN];           // per-batch modulation vector

    const int b    = blockIdx.z;
    const int mt0  = blockIdx.y * BM;        // cout base
    const int y0   = blockIdx.x * 2;         // first of two image rows

    const int tid   = threadIdx.x;
    const int lane  = tid & 31;
    const int wid   = tid >> 5;              // 0..7
    const int waveM = (wid & 3) * 32;        // 4 waves tile 128 couts
    const int waveN = (wid >> 2) * 64;       // 2 waves tile 128 pixels
    const int l15   = lane & 15;
    const int g     = lane >> 4;             // lane half-group per ISA layout

    const float* inb = input + (size_t)b * CIN * H * W;

    // cache style-modulation vector in LDS (2 KB)
    lS[tid]       = smod[(size_t)b * CIN + tid];
    lS[tid + 256] = smod[(size_t)b * CIN + tid + 256];

    v8f acc[2][4];
#pragma unroll
    for (int mi = 0; mi < 2; ++mi)
#pragma unroll
        for (int ni = 0; ni < 4; ++ni)
            acc[mi][ni] = (v8f){0.f, 0.f, 0.f, 0.f, 0.f, 0.f, 0.f, 0.f};

    // A staging: 2 threads per m-row, 16 consecutive cin each
    const int am = tid >> 1;                 // 0..127
    const int ak = (tid & 1) * 16;           // 0 or 16
    const float* wbase = weight + (size_t)(mt0 + am) * (CIN * 9);
    // B staging: thread -> pixel n = tid%128, 16 consecutive cin
    const int bn = tid & 127;
    const int bk = (tid >> 7) * 16;          // 0 or 16
    const int bpy = bn >> 6;                 // row within pixel tile
    const int bx0 = bn & 63;                 // x within row

    float aregs[16];
    float bregs[16];

    // ---- global-load phase for slice s ----
    auto load_slice = [&](int s) {
        const int kk = s >> 4;               // tap 0..8
        const int c0 = (s & 15) * KT;        // cin base
        const int dy = kk / 3 - 1;
        const int dx = kk % 3 - 1;
        // A: weight[mt0+am][c0+ak+j][kh][kw]
#pragma unroll
        for (int j = 0; j < 16; ++j)
            aregs[j] = wbase[(size_t)(c0 + ak + j) * 9 + kk];
        // B: input[b][c0+bk+j][y0+bpy+dy][bx0+dx] with zero pad
        const int yy = y0 + bpy + dy;
        const int xx = bx0 + dx;
        const bool ok = ((unsigned)yy < (unsigned)H) && ((unsigned)xx < (unsigned)W);
        const float* ib = inb + (size_t)yy * W + xx;
#pragma unroll
        for (int j = 0; j < 16; ++j)
            bregs[j] = ok ? ib[(size_t)(c0 + bk + j) * (H * W)] : 0.f;
    };

    // ---- convert + LDS-write phase for slice s into buffer p ----
    auto write_slice = [&](int s, int p) {
        const int c0 = (s & 15) * KT;
        v8bf t0, t1;
#pragma unroll
        for (int j = 0; j < 8; ++j) t0[j] = (__bf16)(aregs[j]     * lS[c0 + ak + j]);
#pragma unroll
        for (int j = 0; j < 8; ++j) t1[j] = (__bf16)(aregs[j + 8] * lS[c0 + ak + 8 + j]);
        *(v8bf*)&lA[p][am * LDK + ak]     = t0;
        *(v8bf*)&lA[p][am * LDK + ak + 8] = t1;
        v8bf u0, u1;
#pragma unroll
        for (int j = 0; j < 8; ++j) u0[j] = (__bf16)bregs[j];
#pragma unroll
        for (int j = 0; j < 8; ++j) u1[j] = (__bf16)bregs[j + 8];
        *(v8bf*)&lB[p][bn * LDK + bk]     = u0;
        *(v8bf*)&lB[p][bn * LDK + bk + 8] = u1;
    };

    // prologue: stage slice 0 into buffer 0 (lS barrier folded in)
    load_slice(0);
    __syncthreads();          // lS ready before write_slice uses it
    write_slice(0, 0);

    for (int s = 0; s < NSLICE; ++s) {
        const int p = s & 1;
        const bool more = (s + 1) < NSLICE;
        if (more) load_slice(s + 1);         // global loads overlap WMMA below
        __syncthreads();                     // buffer p fully written

        // fragments per ISA 16-bit A/B layout: lane holds row l15,
        // K = {g*8..g*8+7} U {16+g*8..16+g*8+7}
        v16bf afrag[2], bfrag[4];
#pragma unroll
        for (int i = 0; i < 2; ++i) {
            const int row = waveM + i * 16 + l15;
            v8bf lo = *(const v8bf*)&lA[p][row * LDK + g * 8];
            v8bf hi = *(const v8bf*)&lA[p][row * LDK + 16 + g * 8];
            afrag[i] = __builtin_shufflevector(lo, hi,
                0,1,2,3,4,5,6,7,8,9,10,11,12,13,14,15);
        }
#pragma unroll
        for (int i = 0; i < 4; ++i) {
            const int col = waveN + i * 16 + l15;
            v8bf lo = *(const v8bf*)&lB[p][col * LDK + g * 8];
            v8bf hi = *(const v8bf*)&lB[p][col * LDK + 16 + g * 8];
            bfrag[i] = __builtin_shufflevector(lo, hi,
                0,1,2,3,4,5,6,7,8,9,10,11,12,13,14,15);
        }
#pragma unroll
        for (int mi = 0; mi < 2; ++mi)
#pragma unroll
            for (int ni = 0; ni < 4; ++ni)
                acc[mi][ni] = __builtin_amdgcn_wmma_f32_16x16x32_bf16(
                    false, afrag[mi], false, bfrag[ni],
                    (short)0, acc[mi][ni], false, false);

        if (more) write_slice(s + 1, p ^ 1); // safe: everyone passed the barrier
    }

    // ---- epilogue: C/D layout M = r + 8*g (VGPR r), N = lane%16 ----
    float* ob = out + ((size_t)b * COUT + mt0) * (H * W) + (size_t)y0 * W;
#pragma unroll
    for (int mi = 0; mi < 2; ++mi)
#pragma unroll
        for (int ni = 0; ni < 4; ++ni) {
            const int n = waveN + ni * 16 + l15;   // 0..127 -> (row, x)
            const int py = n >> 6;
            const int x  = n & 63;
#pragma unroll
            for (int r = 0; r < 8; ++r) {
                const int m = waveM + mi * 16 + g * 8 + r;
                ob[((size_t)m * H + py) * W + x] = acc[mi][ni][r];
            }
        }
}

// ---------------------------------------------------------------------------
extern "C" void kernel_launch(void* const* d_in, const int* in_sizes, int n_in,
                              void* d_out, int out_size, void* d_ws, size_t ws_size,
                              hipStream_t stream) {
    const float* input  = (const float*)d_in[0];   // [8,512,64,64]
    const float* style  = (const float*)d_in[1];   // [8,512]
    const float* weight = (const float*)d_in[2];   // [1,512,512,3,3]
    const float* mod_w  = (const float*)d_in[3];   // [512,512]
    const float* mod_b  = (const float*)d_in[4];   // [512]
    float* out  = (float*)d_out;                   // [8,512,64,64]
    float* smod = (float*)d_ws;                    // [8,512] = 16 KB

    style_mod_kernel<<<(BATCH * CIN) / 256, 256, 0, stream>>>(style, mod_w, mod_b, smod);

    dim3 grid(H / 2, COUT / BM, BATCH);   // 32 x 4 x 8 = 1024 blocks
    modconv_wmma_kernel<<<grid, 256, 0, stream>>>(input, weight, smod, out);
}
